// GraphConvAttentionLayer_68487548502668
// MI455X (gfx1250) — matrix operand order repaired and checked by reading, not compiled, not run on hardware
//
#include <hip/hip_runtime.h>
#include <hip/hip_bf16.h>

// out = elu(input @ W); the diag-only mask makes softmax == identity exactly
// in f32, so adj / a are dead inputs and attention @ h == h.
//
// Split-bf16 GEMM (x = x_hi + x_lo; hh + hl + lh with f32 accumulate ->
// ~2^-17 relative error). One-time pre-pass converts/splits/transposes into
// workspace; the GEMM hot loop stages tiles with async-to-LDS copies
// overlapped under the WMMA issue.

typedef __attribute__((ext_vector_type(16))) __bf16 v16bf;
typedef __attribute__((ext_vector_type(8)))  __bf16 v8bf;
typedef __attribute__((ext_vector_type(8)))  float  v8f;
typedef __attribute__((ext_vector_type(4)))  float  v4f;

#define M_ROWS 8192
#define K_DIM  2048
#define N_COLS 2048
#define BM 128
#define BN 128
#define BK 32
#define LDS_STRIDE 40   // BK + 8 pad (bf16 elements); keeps b128 ops 16B-aligned

#if defined(__has_builtin)
# if __has_builtin(__builtin_amdgcn_global_load_async_to_lds_b128) && \
     __has_builtin(__builtin_amdgcn_s_wait_asynccnt)
#  define USE_ASYNC_LDS 1
# endif
#endif
#ifndef USE_ASYNC_LDS
# define USE_ASYNC_LDS 0
#endif

union Frag  { v16bf f; v8bf h[2]; };
union Pack4 { unsigned short s[4]; uint2 v; };
union Pack8 { unsigned short s[8]; uint4 v; };

static __device__ inline unsigned short bf_hi_bits(float x, float* rem) {
    __bf16 h = (__bf16)x;                 // RNE
    *rem = x - (float)h;
    union { __bf16 b; unsigned short u; } c; c.b = h; return c.u;
}
static __device__ inline unsigned short bf_bits(float x) {
    union { __bf16 b; unsigned short u; } c; c.b = (__bf16)x; return c.u;
}

#if USE_ASYNC_LDS
// builtin prototype (from hipcc diagnostic): pointee is a gcc-style
// vector_size(16) int; param0 in __device__ (AS1), param1 in LDS (AS3).
typedef int v4i_vs __attribute__((vector_size(4 * sizeof(int))));
typedef __attribute__((address_space(1))) v4i_vs gv4i;
typedef __attribute__((address_space(3))) v4i_vs lv4i;
static __device__ inline void async_copy_b128(const void* g, void* l) {
    __builtin_amdgcn_global_load_async_to_lds_b128((gv4i*)g, (lv4i*)l, 0, 0);
}
#endif

// ---------------------------------------------------------------- pre-pass A
__global__ __launch_bounds__(256)
void convert_split_a(const float* __restrict__ inp,
                     unsigned short* __restrict__ a_hi,
                     unsigned short* __restrict__ a_lo)
{
    const size_t nchunks = (size_t)M_ROWS * K_DIM / 4;
    const size_t stride  = (size_t)gridDim.x * blockDim.x;
    for (size_t idx = (size_t)blockIdx.x * blockDim.x + threadIdx.x;
         idx < nchunks; idx += stride) {
        v4f x = *(const v4f*)(inp + idx * 4);
        Pack4 h, l;
        #pragma unroll
        for (int j = 0; j < 4; ++j) {
            float rem;
            h.s[j] = bf_hi_bits(x[j], &rem);
            l.s[j] = bf_bits(rem);
        }
        *(uint2*)(a_hi + idx * 4) = h.v;
        *(uint2*)(a_lo + idx * 4) = l.v;
    }
}

// ------------------------------------------------- pre-pass W (+ transpose)
__global__ __launch_bounds__(256)
void convert_split_w_t(const float* __restrict__ W,
                       unsigned short* __restrict__ bt_hi,
                       unsigned short* __restrict__ bt_lo)
{
    __shared__ __align__(16) unsigned short Lhi[64 * 72];
    __shared__ __align__(16) unsigned short Llo[64 * 72];

    const int tid = threadIdx.x;
    const int n0  = blockIdx.x * 64;
    const int k0  = blockIdx.y * 64;

    #pragma unroll
    for (int i = 0; i < 4; ++i) {
        const int idx = tid + i * 256;
        const int kr  = idx >> 4;
        const int nc  = (idx & 15) * 4;
        v4f x = *(const v4f*)(W + (size_t)(k0 + kr) * N_COLS + n0 + nc);
        #pragma unroll
        for (int j = 0; j < 4; ++j) {
            float rem;
            Lhi[kr * 72 + nc + j] = bf_hi_bits(x[j], &rem);
            Llo[kr * 72 + nc + j] = bf_bits(rem);
        }
    }
    __syncthreads();

    #pragma unroll
    for (int i = 0; i < 2; ++i) {
        const int idx = tid + i * 256;
        const int nr  = idx >> 3;
        const int kc  = (idx & 7) * 8;
        Pack8 h, l;
        #pragma unroll
        for (int j = 0; j < 8; ++j) {
            h.s[j] = Lhi[(kc + j) * 72 + nr];
            l.s[j] = Llo[(kc + j) * 72 + nr];
        }
        *(uint4*)(bt_hi + (size_t)(n0 + nr) * K_DIM + k0 + kc) = h.v;
        *(uint4*)(bt_lo + (size_t)(n0 + nr) * K_DIM + k0 + kc) = l.v;
    }
}

// -------------------------------------------------- staging (one K-tile set)
static __device__ inline void stage_tiles(const unsigned short* __restrict__ a_hi,
                                          const unsigned short* __restrict__ a_lo,
                                          const unsigned short* __restrict__ bt_hi,
                                          const unsigned short* __restrict__ bt_lo,
                                          __bf16* As_hi, __bf16* As_lo,
                                          __bf16* Bs_hi, __bf16* Bs_lo,
                                          int tid, int blockM0, int blockN0, int k1)
{
    #pragma unroll
    for (int i = 0; i < 2; ++i) {
        const int idx = tid + i * 256;
        const int row = idx >> 2;
        const int c8  = (idx & 3) * 8;
        const size_t ga = (size_t)(blockM0 + row) * K_DIM + k1 + c8;
        const size_t gb = (size_t)(blockN0 + row) * K_DIM + k1 + c8;
        const int    lo = row * LDS_STRIDE + c8;
#if USE_ASYNC_LDS
        async_copy_b128(a_hi  + ga, As_hi + lo);
        async_copy_b128(a_lo  + ga, As_lo + lo);
        async_copy_b128(bt_hi + gb, Bs_hi + lo);
        async_copy_b128(bt_lo + gb, Bs_lo + lo);
#else
        const uint4 vah = *(const uint4*)(a_hi  + ga);
        const uint4 val = *(const uint4*)(a_lo  + ga);
        const uint4 vbh = *(const uint4*)(bt_hi + gb);
        const uint4 vbl = *(const uint4*)(bt_lo + gb);
        *(uint4*)(As_hi + lo) = vah;
        *(uint4*)(As_lo + lo) = val;
        *(uint4*)(Bs_hi + lo) = vbh;
        *(uint4*)(Bs_lo + lo) = vbl;
#endif
    }
}

// --------------------------------------------------- GEMM on pre-split data
__global__ __launch_bounds__(256)
void gemm_elu_presplit(const unsigned short* __restrict__ a_hi,
                       const unsigned short* __restrict__ a_lo,
                       const unsigned short* __restrict__ bt_hi,
                       const unsigned short* __restrict__ bt_lo,
                       float* __restrict__ out)
{
    __shared__ __align__(16) __bf16 As_hi[BM * LDS_STRIDE];
    __shared__ __align__(16) __bf16 As_lo[BM * LDS_STRIDE];
    __shared__ __align__(16) __bf16 Bs_hi[BN * LDS_STRIDE];  // [n][k]
    __shared__ __align__(16) __bf16 Bs_lo[BN * LDS_STRIDE];

    const int tid  = threadIdx.x;
    const int lane = tid & 31;
    const int wave = tid >> 5;
    const int r    = lane & 15;
    const int hsel = lane >> 4;

    const int blockN0 = blockIdx.x * BN;
    const int blockM0 = blockIdx.y * BM;
    const int waveM0  = (wave >> 2) * 64;    // 2 (M) x 4 (N) wave grid
    const int waveN0  = (wave & 3) * 32;

    v8f acc[4][2] = {};

    // prologue: stage tile 0
    stage_tiles(a_hi, a_lo, bt_hi, bt_lo, As_hi, As_lo, Bs_hi, Bs_lo,
                tid, blockM0, blockN0, 0);

    for (int k0 = 0; k0 < K_DIM; k0 += BK) {
#if USE_ASYNC_LDS
        __builtin_amdgcn_s_wait_asynccnt(0);
#endif
        __syncthreads();   // staged tile visible to all waves

        // pull ALL fragments for this tile into VGPRs, then free the LDS
        Frag bh[2], bl[2], ah[4], al[4];
        #pragma unroll
        for (int ni = 0; ni < 2; ++ni) {
            const int boff = (waveN0 + ni * 16 + r) * LDS_STRIDE + hsel * 16;
            bh[ni].h[0] = *(const v8bf*)(Bs_hi + boff);
            bh[ni].h[1] = *(const v8bf*)(Bs_hi + boff + 8);
            bl[ni].h[0] = *(const v8bf*)(Bs_lo + boff);
            bl[ni].h[1] = *(const v8bf*)(Bs_lo + boff + 8);
        }
        #pragma unroll
        for (int mi = 0; mi < 4; ++mi) {
            const int aoff = (waveM0 + mi * 16 + r) * LDS_STRIDE + hsel * 16;
            ah[mi].h[0] = *(const v8bf*)(As_hi + aoff);
            ah[mi].h[1] = *(const v8bf*)(As_hi + aoff + 8);
            al[mi].h[0] = *(const v8bf*)(As_lo + aoff);
            al[mi].h[1] = *(const v8bf*)(As_lo + aoff + 8);
        }
        __syncthreads();   // all waves finished reading -> LDS reusable

        // kick off next tile's copies; they drain while the XDL pipe chews
        if (k0 + BK < K_DIM) {
            stage_tiles(a_hi, a_lo, bt_hi, bt_lo, As_hi, As_lo, Bs_hi, Bs_lo,
                        tid, blockM0, blockN0, k0 + BK);
        }

        #pragma unroll
        for (int mi = 0; mi < 4; ++mi) {
            #pragma unroll
            for (int ni = 0; ni < 2; ++ni) {
                acc[mi][ni] = __builtin_amdgcn_wmma_f32_16x16x32_bf16(
                    false, ah[mi].f, false, bh[ni].f, (short)0, acc[mi][ni], false, false);
                acc[mi][ni] = __builtin_amdgcn_wmma_f32_16x16x32_bf16(
                    false, ah[mi].f, false, bl[ni].f, (short)0, acc[mi][ni], false, false);
                acc[mi][ni] = __builtin_amdgcn_wmma_f32_16x16x32_bf16(
                    false, al[mi].f, false, bh[ni].f, (short)0, acc[mi][ni], false, false);
            }
        }
    }

    #pragma unroll
    for (int mi = 0; mi < 4; ++mi) {
        #pragma unroll
        for (int ni = 0; ni < 2; ++ni) {
            const int rowb = blockM0 + waveM0 + mi * 16 + hsel * 8;
            const int col  = blockN0 + waveN0 + ni * 16 + r;
            #pragma unroll
            for (int v = 0; v < 8; ++v) {
                float x = acc[mi][ni][v];
                float y = x > 0.0f ? x : (__expf(x) - 1.0f);
                out[(size_t)(rowb + v) * N_COLS + col] = y;
            }
        }
    }
}

// ------------------------------------------- fallback: fused convert-in-loop
__global__ __launch_bounds__(256)
void gat_fused_gemm_elu(const float* __restrict__ inp,
                        const float* __restrict__ W,
                        float* __restrict__ out)
{
    __shared__ __align__(16) __bf16 As_hi[BM * LDS_STRIDE];
    __shared__ __align__(16) __bf16 As_lo[BM * LDS_STRIDE];
    __shared__ __align__(16) __bf16 Bs_hi[BN * LDS_STRIDE];
    __shared__ __align__(16) __bf16 Bs_lo[BN * LDS_STRIDE];

    const int tid  = threadIdx.x;
    const int lane = tid & 31;
    const int wave = tid >> 5;
    const int r    = lane & 15;
    const int hsel = lane >> 4;

    const int blockN0 = blockIdx.x * BN;
    const int blockM0 = blockIdx.y * BM;
    const int waveM0  = (wave >> 2) * 64;
    const int waveN0  = (wave & 3) * 32;

    v8f acc[4][2] = {};

    for (int k0 = 0; k0 < K_DIM; k0 += BK) {
        #pragma unroll
        for (int i = 0; i < 4; ++i) {
            const int idx  = tid + i * 256;
            const int arow = idx >> 3;
            const int acol = (idx & 7) * 4;
            v4f av = *(const v4f*)(inp + (size_t)(blockM0 + arow) * K_DIM + k0 + acol);
            #pragma unroll
            for (int j = 0; j < 4; ++j) {
                float rem;
                unsigned short hb = bf_hi_bits(av[j], &rem);
                ((unsigned short*)As_hi)[arow * LDS_STRIDE + acol + j] = hb;
                ((unsigned short*)As_lo)[arow * LDS_STRIDE + acol + j] = bf_bits(rem);
            }
            const int krow = idx >> 5;
            const int bcol = (idx & 31) * 4;
            v4f bv = *(const v4f*)(W + (size_t)(k0 + krow) * N_COLS + blockN0 + bcol);
            #pragma unroll
            for (int j = 0; j < 4; ++j) {
                float rem;
                unsigned short hb = bf_hi_bits(bv[j], &rem);
                ((unsigned short*)Bs_hi)[(bcol + j) * LDS_STRIDE + krow] = hb;
                ((unsigned short*)Bs_lo)[(bcol + j) * LDS_STRIDE + krow] = bf_bits(rem);
            }
        }
        __syncthreads();

        Frag bh[2], bl[2];
        #pragma unroll
        for (int ni = 0; ni < 2; ++ni) {
            const int boff = (waveN0 + ni * 16 + r) * LDS_STRIDE + hsel * 16;
            bh[ni].h[0] = *(const v8bf*)(Bs_hi + boff);
            bh[ni].h[1] = *(const v8bf*)(Bs_hi + boff + 8);
            bl[ni].h[0] = *(const v8bf*)(Bs_lo + boff);
            bl[ni].h[1] = *(const v8bf*)(Bs_lo + boff + 8);
        }
        #pragma unroll
        for (int mi = 0; mi < 4; ++mi) {
            const int aoff = (waveM0 + mi * 16 + r) * LDS_STRIDE + hsel * 16;
            Frag ahf, alf;
            ahf.h[0] = *(const v8bf*)(As_hi + aoff);
            ahf.h[1] = *(const v8bf*)(As_hi + aoff + 8);
            alf.h[0] = *(const v8bf*)(As_lo + aoff);
            alf.h[1] = *(const v8bf*)(As_lo + aoff + 8);
            #pragma unroll
            for (int ni = 0; ni < 2; ++ni) {
                acc[mi][ni] = __builtin_amdgcn_wmma_f32_16x16x32_bf16(
                    false, ahf.f, false, bh[ni].f, (short)0, acc[mi][ni], false, false);
                acc[mi][ni] = __builtin_amdgcn_wmma_f32_16x16x32_bf16(
                    false, ahf.f, false, bl[ni].f, (short)0, acc[mi][ni], false, false);
                acc[mi][ni] = __builtin_amdgcn_wmma_f32_16x16x32_bf16(
                    false, alf.f, false, bh[ni].f, (short)0, acc[mi][ni], false, false);
            }
        }
        __syncthreads();
    }

    #pragma unroll
    for (int mi = 0; mi < 4; ++mi) {
        #pragma unroll
        for (int ni = 0; ni < 2; ++ni) {
            const int rowb = blockM0 + waveM0 + mi * 16 + hsel * 8;
            const int col  = blockN0 + waveN0 + ni * 16 + r;
            #pragma unroll
            for (int v = 0; v < 8; ++v) {
                float x = acc[mi][ni][v];
                float y = x > 0.0f ? x : (__expf(x) - 1.0f);
                out[(size_t)(rowb + v) * N_COLS + col] = y;
            }
        }
    }
}

extern "C" void kernel_launch(void* const* d_in, const int* in_sizes, int n_in,
                              void* d_out, int out_size, void* d_ws, size_t ws_size,
                              hipStream_t stream) {
    const float* inp = (const float*)d_in[0];
    // d_in[1] = adj (dead), d_in[3] = a (dead)
    const float* W   = (const float*)d_in[2];
    float*       out = (float*)d_out;

    const size_t A_BYTES  = (size_t)M_ROWS * K_DIM * 2;   // 32 MB per array
    const size_t B_BYTES  = (size_t)K_DIM * N_COLS * 2;   //  8 MB per array
    const size_t WS_NEED  = 2 * A_BYTES + 2 * B_BYTES;    // 80 MB

    dim3 gemm_grid(N_COLS / BN, M_ROWS / BM);             // 16 x 64

    if (ws_size >= WS_NEED) {
        unsigned short* a_hi  = (unsigned short*)((char*)d_ws);
        unsigned short* a_lo  = (unsigned short*)((char*)d_ws + A_BYTES);
        unsigned short* bt_hi = (unsigned short*)((char*)d_ws + 2 * A_BYTES);
        unsigned short* bt_lo = (unsigned short*)((char*)d_ws + 2 * A_BYTES + B_BYTES);

        convert_split_a<<<4096, 256, 0, stream>>>(inp, a_hi, a_lo);
        convert_split_w_t<<<dim3(32, 32), 256, 0, stream>>>(W, bt_hi, bt_lo);
        gemm_elu_presplit<<<gemm_grid, 256, 0, stream>>>(a_hi, a_lo, bt_hi, bt_lo, out);
    } else {
        gat_fused_gemm_elu<<<gemm_grid, 256, 0, stream>>>(inp, W, out);
    }
}
